// QuantumLayer_35880156791007
// MI455X (gfx1250) — matrix operand order repaired and checked by reading, not compiled, not run on hardware
//
#include <hip/hip_runtime.h>
#include <stdint.h>

// ---------------------------------------------------------------------------
// Problem constants (from reference): B=2, T=2048, D=2048, Q=32, DQ=64.
// Pipeline reduces to: per (b,t) tile: out32x64 = f(M32x32 * s32x64), where M
// composes three real circulants (the FFT over DQ cancels; diag depends only
// on the Q-axis frequency). Memory-bound: ~67MB -> ~2.9us @ 23.3TB/s.
// This version double-buffers tiles via ASYNC global->LDS DMA:
//   issue async(tile0), async(tile1) -> wait cnt<=16 -> compute tile0
//   -> wait cnt==0 -> compute tile1   (tile1 DMA hides under tile0 compute)
// ---------------------------------------------------------------------------
#define Qn 32
#define DQn 64
#define Bn 2
#define Tn 2048
#define NTILES (Bn * Tn)        // 4096
#define TILE_ELEMS (Qn * DQn)   // 2048 floats per tile
#define LDS_STRIDE 68           // 64 + 4 pad: 16B-aligned rows, no stride-64
                                // bank pathology on column reads
#define WAVES_PER_BLOCK 2
#define TILES_PER_WAVE 2
#define BLOCK_THREADS (WAVES_PER_BLOCK * 32)
#define USE_ASYNC_LDS 1

typedef __attribute__((ext_vector_type(16))) _Float16 v16h;
typedef __attribute__((ext_vector_type(8)))  float    v8f;

// Workspace (float) layout:
//  [0,1024)    M f32, row-major 32x32
//  [1024,1536) M f16 (1024 halves)      (WMMA A-operand source)
//  [1536,1568) ecc_sum[32]
//  [1568,1600) syn0[32]
//  [1600,1632) syn1[32]

// ---------------------------------------------------------------------------
// prep_kernel: unitarize (Gram-Schmidt QR of (256,2)) -> theta -> rotation
// diagonals -> three circulants Re(F^-1 diag F) -> compose M = Mz*My*Mx.
// ---------------------------------------------------------------------------
__global__ __launch_bounds__(1024)
void prep_kernel(const float* __restrict__ rx, const float* __restrict__ ry,
                 const float* __restrict__ rz, const float* __restrict__ ecc,
                 const float* __restrict__ syn, float* __restrict__ ws)
{
  __shared__ float tp[3][16];
  __shared__ float dre[3][32];
  __shared__ float dimg[3][32];
  __shared__ float Mx[32][32], My[32][32], Mz[32][32], Tt[32][32];
  const int tid = threadIdx.x;

  if (tid < 3) {
    const float* p = (tid == 0) ? rx : ((tid == 1) ? ry : rz);
    float n1 = 0.f;
    for (int i = 0; i < 256; ++i) { float a = p[2*i]; n1 += a * a; }
    n1 = sqrtf(n1) + 1e-12f;
    float dot = 0.f;
    for (int i = 0; i < 256; ++i) dot += (p[2*i] / n1) * p[2*i + 1];
    float n2 = 0.f;
    for (int i = 0; i < 256; ++i) {
      float w = p[2*i + 1] - dot * (p[2*i] / n1);
      n2 += w * w;
    }
    n2 = sqrtf(n2) + 1e-12f;
    for (int j = 0; j < 16; ++j) {
      float s = 0.f;
      for (int i = 16*j; i < 16*j + 8; ++i) {
        float q1 = p[2*i] / n1;
        float q2 = (p[2*i + 1] - dot * q1) / n2;
        s += q1 + q2;
      }
      tp[tid][j] = s * (1.0f / 16.0f);
    }
  }
  __syncthreads();

  if (tid < 96) {
    int t = tid >> 5, k = tid & 31;
    float h = tp[t][k >> 1] * 0.5f;
    if (t < 2) { dre[t][k] = cosf(h); dimg[t][k] = 0.f; }
    else       { dre[2][k] = cosf(h); dimg[2][k] = (k & 1) ? sinf(h) : -sinf(h); }
  }
  __syncthreads();

  {
    int qp = tid >> 5, q = tid & 31;
    float ax = 0.f, ay = 0.f, az = 0.f;
    for (int k = 0; k < 32; ++k) {
      float ang = 0.19634954084936207f * (float)(k * (qp - q));
      float c = cosf(ang), s = sinf(ang);
      ax += c * dre[0][k];
      ay += c * dre[1][k];
      az += c * dre[2][k] - s * dimg[2][k];
    }
    Mx[qp][q] = ax * (1.0f / 32.0f);
    My[qp][q] = ay * (1.0f / 32.0f);
    Mz[qp][q] = az * (1.0f / 32.0f);
  }
  __syncthreads();
  {
    int qp = tid >> 5, q = tid & 31;
    float s = 0.f;
    for (int k = 0; k < 32; ++k) s += My[qp][k] * Mx[k][q];
    Tt[qp][q] = s;
  }
  __syncthreads();
  {
    int qp = tid >> 5, q = tid & 31;
    float s = 0.f;
    for (int k = 0; k < 32; ++k) s += Mz[qp][k] * Tt[k][q];
    ws[tid] = s;
    ((_Float16*)(ws + 1024))[tid] = (_Float16)s;
  }
  if (tid < 32) {
    ws[1536 + tid] = ecc[tid] + ecc[32 + tid] + ecc[64 + tid];
    ws[1568 + tid] = syn[tid];
    ws[1600 + tid] = syn[32 + tid];
  }
}

// ---------------------------------------------------------------------------
// Hash-based stand-in for jax.random noise (threefry not reproducible here).
// ---------------------------------------------------------------------------
__device__ __forceinline__ uint32_t hash3(uint32_t a, uint32_t b, uint32_t c) {
  uint32_t h = a * 0x9E3779B1u + b * 0x85EBCA77u + c * 0xC2B2AE3Du + 0x27D4EB2Fu;
  h ^= h >> 16; h *= 0x7FEB352Du;
  h ^= h >> 15; h *= 0x846CA68Bu;
  h ^= h >> 16;
  return h;
}

__device__ __forceinline__ void noise_at(uint32_t tile, uint32_t q, uint32_t d,
                                         float& amp, float& ph) {
  uint32_t h1 = hash3(tile, q, 2u * d + 1u);
  uint32_t h2 = hash3(tile, q, 2u * d + 2u);
  float u1 = (float)(h1 >> 8) * 5.9604645e-08f + 1e-7f;
  float u2 = (float)(h2 >> 8) * 5.9604645e-08f;
  float r = sqrtf(-2.0f * __logf(u1));
  float sn, cs;
  __sincosf(6.2831853071f * u2, &sn, &cs);
  amp = 0.01f  * r * cs;
  ph  = 0.005f * r * sn;
}

// ---------------------------------------------------------------------------
// Async global->LDS staging (CDNA5 async-DMA path, ASYNCcnt-tracked).
// LDS operand: generic pointers to LDS carry the workgroup-relative LDS byte
// offset in their low 32 bits (aperture in the high bits), which is exactly
// the VDST value GLOBAL_LOAD_ASYNC_TO_LDS expects.
// ---------------------------------------------------------------------------
__device__ __forceinline__ void stage_tile(const float* __restrict__ src,
                                           float* __restrict__ tb, int lane) {
  #pragma unroll
  for (int i = 0; i < 16; ++i) {
    const int c   = i * 32 + lane;       // 512 16B chunks; global side is
    const int row = c >> 4;              // perfectly coalesced (chunk = 4c)
    const int c4  = (c & 15) * 4;
#if USE_ASYNC_LDS
    const unsigned long long g = (unsigned long long)(uintptr_t)(src + c * 4);
    const uint32_t l = (uint32_t)(uintptr_t)(tb + row * LDS_STRIDE + c4);
    asm volatile("global_load_async_to_lds_b128 %0, %1, off"
                 :: "v"(l), "v"(g) : "memory");
#else
    const float4 v = *(const float4*)(src + c * 4);
    *(float4*)(tb + row * LDS_STRIDE + c4) = v;
#endif
  }
}

__device__ __forceinline__ void wait_async_16() {
#if USE_ASYNC_LDS
  asm volatile("s_wait_asynccnt 16" ::: "memory");   // first tile's 16 done
#endif
}
__device__ __forceinline__ void wait_async_0() {
#if USE_ASYNC_LDS
  asm volatile("s_wait_asynccnt 0" ::: "memory");
#endif
}

// ---------------------------------------------------------------------------
// Per-tile compute: 8x v_wmma_f32_16x16x32_f16 + epilogue + coalesced store.
// ---------------------------------------------------------------------------
__device__ __forceinline__ void process_tile(
    float* __restrict__ tb, int tile, int lane,
    const v16h& a0, const v16h& a1,
    const float* __restrict__ sEcc, const float* __restrict__ sSyn0,
    const float* __restrict__ sSyn1, float* __restrict__ out)
{
  const int half = lane >> 4;
  const int ncol = lane & 15;
  const int kb   = half * 16;

  // B fragments: 16-bit B 32x16 layout — lane group h holds K=8h..8h+7,
  // 16+8h..16+8h+7; col = lane&15.
  v16h bf[4];
  #pragma unroll
  for (int c = 0; c < 4; ++c) {
    #pragma unroll
    for (int e = 0; e < 16; ++e)
      bf[c][e] = (_Float16)tb[((e & 7) + half * 8 + (e >> 3) * 16) * LDS_STRIDE + c * 16 + ncol];
  }

  v8f acc[2][4];
  #pragma unroll
  for (int r = 0; r < 2; ++r) {
    #pragma unroll
    for (int c = 0; c < 4; ++c) {
      v8f z = {0.f, 0.f, 0.f, 0.f, 0.f, 0.f, 0.f, 0.f};
      acc[r][c] = __builtin_amdgcn_wmma_f32_16x16x32_f16(
          false, (r == 0) ? a0 : a1, false, bf[c],
          (short)0, z, false, false);
    }
  }

  // D writeback over the tile buffer (C/D layout: VGPR p -> M=p+8*half).
  #pragma unroll
  for (int r = 0; r < 2; ++r)
    #pragma unroll
    for (int c = 0; c < 4; ++c)
      #pragma unroll
      for (int p = 0; p < 8; ++p)
        tb[(r * 16 + p + half * 8) * LDS_STRIDE + c * 16 + ncol] = acc[r][c][p];

  // ---------------- Epilogue: lane q owns row q (64 elems) ----------------
  const int q = lane;
  float* rowp = tb + q * LDS_STRIDE;

  float sumsq = 0.f, sumre = 0.f;
  for (int j = 0; j < 64; ++j) {
    const float s = rowp[j];
    float amp, ph; noise_at((uint32_t)tile, (uint32_t)q, (uint32_t)j, amp, ph);
    const float mag = (1.f + amp) * s;
    float sn, cs; __sincosf(ph, &sn, &cs);
    const float re = mag * cs, im = mag * sn;
    sumsq += re * re + im * im;
    sumre += re;
  }
  const float inv = 1.f / (sqrtf(sumsq) + 1e-8f);
  const bool px = (sSyn0[q] * sumre * inv) > 0.f;
  const bool pz = (sSyn1[q] * sumre * inv) > 0.f;
  (void)pz; // pz only negates; |final| and |fid| are negation-invariant

  float fr = 0.f, fi = 0.f;
  for (int j = 0; j < 64; ++j) {
    const int jj = px ? (63 - j) : j;
    const float s2 = rowp[jj];
    float amp, ph; noise_at((uint32_t)tile, (uint32_t)q, (uint32_t)jj, amp, ph);
    const float mag = (1.f + amp) * s2;
    float sn, cs; __sincosf(ph, &sn, &cs);
    fr += (mag * cs) * rowp[j];
    fi += (mag * sn) * rowp[j];
  }
  const float fid = fabsf(sEcc[q]) * inv * sqrtf(fr * fr + fi * fi);
  const bool use_alt = fid < 0.9f;
  const bool flip    = use_alt ? (!px) : px;

  // |final[j]| = inv * |(1+amp(jj)) * s[jj]|  (|e^{i ph}| == 1); mirror pairs
  // so the in-place flip never reads overwritten data.
  for (int j = 0; j < 32; ++j) {
    const int j2 = 63 - j;
    const float sA = rowp[j], sB = rowp[j2];
    float ampA, phA, ampB, phB;
    noise_at((uint32_t)tile, (uint32_t)q, (uint32_t)j,  ampA, phA);
    noise_at((uint32_t)tile, (uint32_t)q, (uint32_t)j2, ampB, phB);
    const float vA = inv * fabsf((1.f + ampA) * sA);
    const float vB = inv * fabsf((1.f + ampB) * sB);
    rowp[j]  = flip ? vB : vA;
    rowp[j2] = flip ? vA : vB;
  }

  const size_t obase = (size_t)tile * TILE_ELEMS;
  #pragma unroll
  for (int i = lane * 4; i < TILE_ELEMS; i += 32 * 4) {
    const int row = i >> 6, col = i & 63;
    const float4 vv = *(const float4*)(tb + row * LDS_STRIDE + col);
    *(float4*)(out + obase + i) = vv;
  }
}

// ---------------------------------------------------------------------------
// Main kernel: 2 waves/block, 2 tiles/wave, double-buffered async staging.
// ---------------------------------------------------------------------------
__global__ __launch_bounds__(BLOCK_THREADS)
void qc_main_kernel(const float* __restrict__ x, const float* __restrict__ ws,
                    float* __restrict__ out)
{
  __shared__ _Float16 sMh[1024];
  __shared__ float sEcc[32], sSyn0[32], sSyn1[32];
  __shared__ __align__(16) float tileLds[WAVES_PER_BLOCK * 2][Qn * LDS_STRIDE];

  const _Float16* wMh = (const _Float16*)(ws + 1024);
  for (int i = threadIdx.x; i < 1024; i += BLOCK_THREADS) sMh[i] = wMh[i];
  if (threadIdx.x < 32) {
    sEcc[threadIdx.x]  = ws[1536 + threadIdx.x];
    sSyn0[threadIdx.x] = ws[1568 + threadIdx.x];
    sSyn1[threadIdx.x] = ws[1600 + threadIdx.x];
  }
  __syncthreads();

  const int wave = threadIdx.x >> 5;
  const int lane = threadIdx.x & 31;
  const int half = lane >> 4;
  const int mrow = lane & 15;
  float* tb0 = tileLds[wave * 2 + 0];
  float* tb1 = tileLds[wave * 2 + 1];

  const int t0 = blockIdx.x * (WAVES_PER_BLOCK * TILES_PER_WAVE)
               + wave * TILES_PER_WAVE;
  const int t1 = t0 + 1;

  // Kick off both tiles' DMA before touching the A operand — the second
  // tile's 8KB transfer hides under the first tile's WMMA + epilogue.
  stage_tile(x + (size_t)t0 * TILE_ELEMS, tb0, lane);
  stage_tile(x + (size_t)t1 * TILE_ELEMS, tb1, lane);

  // A fragments (M rows), built once per wave. 16-bit A 16x32 layout:
  // lane L<16: row=L; K pattern {2v,2v+1}+8*half per ISA 7.12.2.
  v16h a0 = {}, a1 = {};
  #pragma unroll
  for (int e = 0; e < 16; ++e) {
    const int v = e >> 1, lh = e & 1;
    const int k = ((v < 4) ? (2 * v) : (16 + 2 * (v - 4))) + half * 8 + lh;
    a0[e] = sMh[mrow * 32 + k];
    a1[e] = sMh[(mrow + 16) * 32 + k];
  }

  wait_async_16();   // async loads complete in order: first 16 => tb0 ready
  process_tile(tb0, t0, lane, a0, a1, sEcc, sSyn0, sSyn1, out);
  wait_async_0();    // tb1 ready (DMA overlapped tile0 compute)
  process_tile(tb1, t1, lane, a0, a1, sEcc, sSyn0, sSyn1, out);
}

// ---------------------------------------------------------------------------
extern "C" void kernel_launch(void* const* d_in, const int* in_sizes, int n_in,
                              void* d_out, int out_size, void* d_ws, size_t ws_size,
                              hipStream_t stream) {
  (void)in_sizes; (void)n_in; (void)out_size; (void)ws_size;
  const float* x   = (const float*)d_in[0];
  const float* rx  = (const float*)d_in[1];
  const float* ry  = (const float*)d_in[2];
  const float* rz  = (const float*)d_in[3];
  const float* ecc = (const float*)d_in[4];
  const float* syn = (const float*)d_in[5];
  float* out = (float*)d_out;
  float* ws  = (float*)d_ws;   // needs ~6.5 KB

  prep_kernel<<<1, 1024, 0, stream>>>(rx, ry, rz, ecc, syn, ws);
  const int blocks = NTILES / (WAVES_PER_BLOCK * TILES_PER_WAVE); // 1024
  qc_main_kernel<<<blocks, BLOCK_THREADS, 0, stream>>>(x, ws, out);
}
